// GaussianMixture_6116033429850
// MI455X (gfx1250) — matrix operand order repaired and checked by reading, not compiled, not run on hardware
//
#include <hip/hip_runtime.h>
#include <hip/hip_bf16.h>
#include <math.h>

#define DIMS 20
#define KCOMP 168
#define KPAD 176          // 11 tiles of 16 components
#define NTILES 11
#define KIN 64            // inner dim padded 41 -> 64 (two k=32 WMMA steps)
#define ROWS_PER_WAVE 16
#define WAVES_PER_BLOCK 8
#define ROWS_PER_BLOCK 128
#define WS_B_OFF 1024

#define LOG_NORM_F (-18.3787706641f)        // -DIMS/2 * ln(2*pi)
#define NEG_HALF_LOG2E (-0.72134752044f)    // -0.5 * log2(e)
#define LOG2E_F (1.44269504089f)
#define LN2_F (0.69314718056f)

typedef __attribute__((ext_vector_type(16))) __bf16 v16bf;
typedef __attribute__((ext_vector_type(8)))  float  v8f;

union V16 { int4 q[2]; v16bf v; };

__device__ __forceinline__ unsigned short f2bf(float f) {
  unsigned int u = __float_as_uint(f);
  u += 0x7FFFu + ((u >> 16) & 1u);   // round-to-nearest-even to bf16
  return (unsigned short)(u >> 16);
}

// ---------------------------------------------------------------------------
// Stage 1: softmax, per-component log2(weight*sig_term), and the B matrix
// written directly in WMMA B-register image order (bf16).
// B row-features: [inv(0..19), -2*mu*inv(20..39), sum(mu^2*inv)(40), 0...]
// ---------------------------------------------------------------------------
__global__ void gm_prep(const float* __restrict__ alpha,
                        const float* __restrict__ mu,
                        const float* __restrict__ cov,
                        float* __restrict__ lw,
                        unsigned short* __restrict__ Bimg)
{
  __shared__ float red[256];
  const int t = threadIdx.x;
  const float var = 1.0f;  // EPS^2 with EPS=1.0

  float a = (t < KCOMP) ? alpha[t] : -INFINITY;
  red[t] = a; __syncthreads();
  for (int s = 128; s > 0; s >>= 1) {
    if (t < s) red[t] = fmaxf(red[t], red[t + s]);
    __syncthreads();
  }
  float amax = red[0]; __syncthreads();
  float e = (t < KCOMP) ? __builtin_amdgcn_exp2f((a - amax) * LOG2E_F) : 0.f;
  red[t] = e; __syncthreads();
  for (int s = 128; s > 0; s >>= 1) {
    if (t < s) red[t] += red[t + s];
    __syncthreads();
  }
  float denom = red[0];

  if (t < KPAD) {
    float v = -INFINITY;
    if (t < KCOMP) {
      float w   = e / denom;
      float fi  = (float)(t >> 3);                       // i = k // MULT
      float det = powf(var, fi) * powf(0.1f, (float)DIMS - fi);
      float sig = powf(det, -0.5f);
      v = __builtin_amdgcn_logf(w * sig);                // v_log_f32 = log2
    }
    lw[t] = v;
  }

  for (int idx = t; idx < KIN * KPAD; idx += 256) {
    int comp = idx % KPAD;
    int K    = idx / KPAD;
    float val = 0.f;
    if (comp < KCOMP) {
      int ig = comp >> 3;
      if (K < DIMS) {
        float s = (K < ig) ? var : 1.f;
        val = 1.f / (cov[comp * DIMS + K] * s);
      } else if (K < 2 * DIMS) {
        int d = K - DIMS;
        float s = (d < ig) ? var : 1.f;
        float inv = 1.f / (cov[comp * DIMS + d] * s);
        val = -2.f * mu[comp * DIMS + d] * inv;
      } else if (K == 2 * DIMS) {
        float c = 0.f;
        for (int d = 0; d < DIMS; ++d) {
          float s = (d < ig) ? var : 1.f;
          float inv = 1.f / (cov[comp * DIMS + d] * s);
          float m = mu[comp * DIMS + d];
          c += m * m * inv;
        }
        val = c;
      }
    }
    // Map (K, comp) -> B register image slot.
    // B 32x16 in this code's contiguous k order (A matches): lanes 0-15 column N hold K=0..15 (v0.lo..v7.hi),
    // lanes 16-31 hold K=16..31.
    int tIdx = comp >> 4, nl = comp & 15;
    int b  = K >> 5, kk = K & 31;
    int lane = nl + ((kk >= 16) ? 16 : 0);
    int slot = kk & 15;                    // sequential bf16 slot in lane's 32B
    int ui = (((tIdx * 2 + b) * 32) + lane) * 16 + slot;
    Bimg[ui] = f2bf(val);
  }
}

// ---------------------------------------------------------------------------
// Stage 2: main WMMA kernel. 8 waves/block, 16 rows/wave.
// ---------------------------------------------------------------------------
__global__ void __launch_bounds__(256) gm_main(
    const float* __restrict__ sample,
    const float* __restrict__ lw,
    const int4*  __restrict__ Bq,     // B register image, 16B granular
    float* __restrict__ out,
    int N)
{
  __shared__ __align__(16) unsigned short feat[WAVES_PER_BLOCK][ROWS_PER_WAVE][KIN];
  const int lane = threadIdx.x & 31;
  const int wave = threadIdx.x >> 5;
  const int rowBase = blockIdx.x * ROWS_PER_BLOCK + wave * ROWS_PER_WAVE;
  const bool active = (rowBase + ROWS_PER_WAVE) <= N;

  // Phase 1: build bf16 feature rows [x^2(20), x(20), 1, 0...] in LDS.
  if (active) {
    for (int r = 0; r < ROWS_PER_WAVE; ++r) {
      int n = rowBase + r;
      unsigned int* fd = (unsigned int*)feat[wave][r];
      if (lane < 10) {
        int d = lane * 2;
        float2 x = *(const float2*)(sample + (size_t)n * DIMS + d);
        unsigned int sq = (unsigned int)f2bf(x.x * x.x) |
                          ((unsigned int)f2bf(x.y * x.y) << 16);
        unsigned int ln = (unsigned int)f2bf(x.x) |
                          ((unsigned int)f2bf(x.y) << 16);
        fd[lane]      = sq;   // K = 2*lane, 2*lane+1
        fd[lane + 10] = ln;   // K = 2*lane+20, +21
      } else if (lane == 10) {
        fd[20] = (unsigned int)f2bf(1.0f);   // K=40 constant, K=41 zero
      } else if (lane <= 21) {
        fd[lane + 10] = 0u;                  // K=42..63 zero padding
      }
    }
  }
  __syncthreads();
  if (!active) return;

  // Phase 2: A fragments (16x32 per step) in contiguous k order, matching B.
  const int r  = lane & 15;
  const int hs = (lane >> 4) & 1;   // lanes 16-31 hold K 16..31
  const unsigned short* fr = feat[wave][r];

  V16 A0, A1;
  A0.q[0] = *(const int4*)(fr +  0 + hs * 16);
  A0.q[1] = *(const int4*)(fr +  8 + hs * 16);
  A1.q[0] = *(const int4*)(fr + 32 + hs * 16);
  A1.q[1] = *(const int4*)(fr + 40 + hs * 16);

  float dens[8];
  #pragma unroll
  for (int j = 0; j < 8; ++j) dens[j] = 0.f;

  #pragma unroll
  for (int t = 0; t < NTILES; ++t) {
    V16 B0, B1;
    int bi = ((t * 2 + 0) * 32 + lane) * 2;
    B0.q[0] = Bq[bi];     B0.q[1] = Bq[bi + 1];
    int bj = ((t * 2 + 1) * 32 + lane) * 2;
    B1.q[0] = Bq[bj];     B1.q[1] = Bq[bj + 1];

    v8f acc = {};
    acc = __builtin_amdgcn_wmma_f32_16x16x32_bf16(false, A0.v, false, B0.v,
                                                  (short)0, acc, false, false);
    acc = __builtin_amdgcn_wmma_f32_16x16x32_bf16(false, A1.v, false, B1.v,
                                                  (short)0, acc, false, false);
    float lwv = lw[t * 16 + r];   // log2(weight*sig) for this lane's component
    #pragma unroll
    for (int j = 0; j < 8; ++j)
      dens[j] += __builtin_amdgcn_exp2f(fmaf(acc[j], NEG_HALF_LOG2E, lwv));
  }

  // Reduce over the 16 components held across each 16-lane half.
  #pragma unroll
  for (int j = 0; j < 8; ++j) {
    dens[j] += __shfl_xor(dens[j], 1);
    dens[j] += __shfl_xor(dens[j], 2);
    dens[j] += __shfl_xor(dens[j], 4);
    dens[j] += __shfl_xor(dens[j], 8);
  }

  if (r == 0) {
    // lane 0 holds rows rowBase+0..7, lane 16 holds rows rowBase+8..15
    float* o = out + rowBase + hs * 8;
    #pragma unroll
    for (int j = 0; j < 8; ++j)
      o[j] = __builtin_amdgcn_logf(dens[j]) * LN2_F + LOG_NORM_F;
  }
}

extern "C" void kernel_launch(void* const* d_in, const int* in_sizes, int n_in,
                              void* d_out, int out_size, void* d_ws, size_t ws_size,
                              hipStream_t stream) {
  const float* sample = (const float*)d_in[0];
  const float* alpha  = (const float*)d_in[1];
  const float* mu     = (const float*)d_in[2];
  const float* cov    = (const float*)d_in[3];
  float* out = (float*)d_out;
  int N = in_sizes[0] / DIMS;

  float* lw = (float*)d_ws;
  unsigned short* Bimg = (unsigned short*)((char*)d_ws + WS_B_OFF);

  hipLaunchKernelGGL(gm_prep, dim3(1), dim3(256), 0, stream,
                     alpha, mu, cov, lw, Bimg);

  int blocks = (N + ROWS_PER_BLOCK - 1) / ROWS_PER_BLOCK;
  hipLaunchKernelGGL(gm_main, dim3(blocks), dim3(256), 0, stream,
                     sample, lw, (const int4*)Bimg, out, N);
}
